// RoPESelfAttention_77163382440664
// MI455X (gfx1250) — compile-verified, matrix operand order checked by reading, not run on hardware
//
#include <hip/hip_runtime.h>

#define B_ 8
#define S_ 2048
#define H_ 256
#define NROW_ (B_ * S_)   // 16384 rows of the flattened [B*S, H] activation matrix

typedef __attribute__((ext_vector_type(16))) __bf16 bf16x16;
typedef __attribute__((ext_vector_type(8)))  float  f32x8;
typedef __attribute__((ext_vector_type(4)))  int    i32x4;

#if defined(__has_builtin)
#if __has_builtin(__builtin_amdgcn_global_load_async_to_lds_b128)
#define HAVE_ASYNC_LDS 1
#endif
#endif
#ifndef HAVE_ASYNC_LDS
#define HAVE_ASYNC_LDS 0
#endif

// ---------- helpers ----------

__device__ __forceinline__ unsigned short f2bf(float f) {
  unsigned int u = __float_as_uint(f);
  return (unsigned short)((u + 0x7FFFu + ((u >> 16) & 1u)) >> 16);  // RNE
}

// Load a 16-element bf16 fragment for WMMA A/B operands.
// rowk0 points at (row, k0) of a row-major bf16 matrix. c0 = (lane>=16)*8.
// elements 0..7  <- K = c0 .. c0+7 ; elements 8..15 <- K = 16+c0 .. 16+c0+7
__device__ __forceinline__ bf16x16 load_frag(const unsigned short* rowk0, int c0) {
  union { unsigned int u[8]; bf16x16 v; } r;
  const unsigned int* p0 = (const unsigned int*)(rowk0 + c0);
  const unsigned int* p1 = (const unsigned int*)(rowk0 + c0 + 16);
#pragma unroll
  for (int i = 0; i < 4; ++i) { r.u[i] = p0[i]; r.u[4 + i] = p1[i]; }
  return r.v;
}

__device__ __forceinline__ f32x8 wmma_bf16(bf16x16 a, bf16x16 b, f32x8 c) {
  return __builtin_amdgcn_wmma_f32_16x16x32_bf16(false, a, false, b, (short)0, c, false, false);
}

#if HAVE_ASYNC_LDS
typedef __attribute__((address_space(1))) i32x4* gv4_t;   // global <4 x i32>*
typedef __attribute__((address_space(3))) i32x4* lv4_t;   // LDS    <4 x i32>*

__device__ __forceinline__ void cp16_async(const void* g, void* l) {
  __builtin_amdgcn_global_load_async_to_lds_b128((gv4_t)(unsigned long)g,
                                                 (lv4_t)l, 0, 0);
}
__device__ __forceinline__ void wait_async0() {
#if __has_builtin(__builtin_amdgcn_s_wait_asynccnt)
  __builtin_amdgcn_s_wait_asynccnt(0);
#else
  asm volatile("s_wait_asynccnt 0x0" ::: "memory");
#endif
}
#endif

// ---------- kernel 1: weights -> bf16 ----------

__global__ void cvt_weights_k(const float* __restrict__ wq,
                              const float* __restrict__ wk,
                              const float* __restrict__ wv,
                              unsigned short* __restrict__ out) {
  int i = blockIdx.x * 256 + threadIdx.x;          // 0 .. 3*65536-1
  const float* src = (i < 65536) ? wq : (i < 131072 ? wk : wv);
  out[i] = f2bf(src[i & 65535]);
}

// ---------- kernel 2: extract RoPE cos/sin from dense rotary buffer ----------

__global__ void build_cs_k(const float* __restrict__ r, float2* __restrict__ cs) {
  int s = blockIdx.x;                               // 0..2047
  int i = threadIdx.x;                              // 0..127 (pair index)
  const float* rp = r + (size_t)s * H_ * H_;
  float c  = rp[(size_t)(2 * i) * H_ + 2 * i];      // R[s, 2i,   2i] = cos
  float sn = rp[(size_t)(2 * i + 1) * H_ + 2 * i];  // R[s, 2i+1, 2i] = sin
  cs[s * 128 + i] = make_float2(c, sn);
}

// ---------- kernel 3: QKV projection + RoPE ----------
// Block = 128 threads (4 waves). Each block owns 16 consecutive rows of [B*S, H].

__global__ void proj_rope_k(const float* __restrict__ x,
                            const unsigned short* __restrict__ wbf,   // wq|wk|wv bf16
                            const float2* __restrict__ cs,
                            unsigned short* __restrict__ qr,
                            unsigned short* __restrict__ kr,
                            unsigned short* __restrict__ vT) {
  __shared__ __align__(16) unsigned short xs[16 * 256];   // x tile, bf16
  __shared__ __align__(16) float qk[16 * 256];            // fp32 staging for RoPE

  const int tile = blockIdx.x;              // 0..1023
  const int row0 = tile * 16;               // base row in [B*S, H]
  const int tid  = threadIdx.x;
  const int lane = tid & 31, wave = tid >> 5;
  const int half = lane >> 4, lr = lane & 15, c0 = half * 8;
  const int b = row0 / S_, s_in = row0 % S_;  // 16-row tile never straddles batches

  for (int idx = tid; idx < 16 * 256; idx += 128) {
    int rr = idx >> 8, cc = idx & 255;
    xs[idx] = f2bf(x[(size_t)(row0 + rr) * H_ + cc]);
  }
  __syncthreads();

  for (int mat = 0; mat < 3; ++mat) {
    const unsigned short* w = wbf + mat * 65536;
    f32x8 acc[4];
#pragma unroll
    for (int t = 0; t < 4; ++t) {
      int nt = wave * 4 + t;                // output N tile (0..15)
      f32x8 a = {};
#pragma unroll
      for (int kk = 0; kk < 8; ++kk) {
        bf16x16 af = load_frag(&xs[lr * 256 + kk * 32], c0);
        bf16x16 bf = load_frag(&w[(size_t)(nt * 16 + lr) * 256 + kk * 32], c0);
        a = wmma_bf16(af, bf, a);
      }
      acc[t] = a;
    }

    if (mat < 2) {
      // stage fp32 result to LDS, then apply RoPE with per-element mapping
#pragma unroll
      for (int t = 0; t < 4; ++t) {
        int nt = wave * 4 + t;
#pragma unroll
        for (int e = 0; e < 8; ++e)
          qk[(e + half * 8) * 256 + nt * 16 + lr] = acc[t][e];
      }
      __syncthreads();
      unsigned short* dst = (mat == 0) ? qr : kr;
      for (int idx = tid; idx < 16 * 128; idx += 128) {
        int rr = idx >> 7, i = idx & 127;
        float2 csv = cs[(s_in + rr) * 128 + i];
        float q0 = qk[rr * 256 + 2 * i];
        float q1 = qk[rr * 256 + 2 * i + 1];
        float o0 =  csv.x * q0 + csv.y * q1;
        float o1 = -csv.y * q0 + csv.x * q1;
        size_t go = (size_t)(row0 + rr) * H_ + 2 * i;
        dst[go]     = f2bf(o0);
        dst[go + 1] = f2bf(o1);
      }
      __syncthreads();
    } else {
      // v: store transposed vT[b][h][s] for contiguous PV B-fragments
#pragma unroll
      for (int t = 0; t < 4; ++t) {
        int nt = wave * 4 + t;
        int h = nt * 16 + lr;
#pragma unroll
        for (int e = 0; e < 8; ++e) {
          int rr = e + half * 8;
          vT[((size_t)b * H_ + h) * S_ + (s_in + rr)] = f2bf(acc[t][e]);
        }
      }
    }
  }
}

// ---------- kernel 4: causal flash attention ----------
// Block = 4 waves; each wave owns one 16-query tile (block = 64 queries).
// K/V tiles (32 keys) are staged once into LDS per block and shared by all 4
// waves -> 4x less L2 read traffic. Staging uses async global->LDS copies
// (ASYNCcnt + s_wait_asynccnt) when available.

__global__ void attn_k(const unsigned short* __restrict__ qr,
                       const unsigned short* __restrict__ kr,
                       const unsigned short* __restrict__ vT,
                       float* __restrict__ out) {
  __shared__ __align__(16) unsigned short kbuf[32 * 256];   // [key in tile][h] 16 KB
  __shared__ __align__(16) unsigned short vbuf[256 * 32];   // [h][key in tile] 16 KB
  __shared__ __align__(16) unsigned short pbuf[4][16 * 32]; // per-wave P tile   4 KB

  const int blk  = blockIdx.x;               // 256 blocks
  const int b    = blk >> 5;                 // 32 blocks per batch
  const int qt   = (blk & 31) * 4 + (threadIdx.x >> 5);  // this wave's q tile
  const int tid  = threadIdx.x;
  const int wave = tid >> 5;
  const int lane = tid & 31;
  const int half = lane >> 4, lr = lane & 15, c0 = half * 8;

  // resident Q fragments: row = query (lane&15), K = H
  bf16x16 af[8];
  const size_t qrow = (size_t)(b * S_ + qt * 16 + lr) * H_;
#pragma unroll
  for (int kk = 0; kk < 8; ++kk) af[kk] = load_frag(&qr[qrow + kk * 32], c0);

  f32x8 acc[16];
#pragma unroll
  for (int nt = 0; nt < 16; ++nt) acc[nt] = (f32x8){};
  float m_st[8], l_st[8];
#pragma unroll
  for (int e = 0; e < 8; ++e) { m_st[e] = -1e30f; l_st[e] = 0.0f; }

  const int jmax_self = (qt * 16 + 15) >> 5;
  const int jmax_blk  = ((((blk & 31) * 4 + 3) * 16) + 15) >> 5;  // block-uniform

  for (int j = 0; j <= jmax_blk; ++j) {
    __syncthreads();  // previous iteration's LDS reads complete before overwrite

    // ---- cooperative stage of kr / vT tiles into LDS (16B per thread-iter) ----
    {
      const uint4* krg = (const uint4*)(kr + (size_t)(b * S_ + j * 32) * H_);  // contiguous 16KB
      uint4* kbl = (uint4*)kbuf;
      uint4* vbl = (uint4*)vbuf;
#pragma unroll
      for (int it = 0; it < 8; ++it) {
        int idx = tid + it * 128;               // uint4 index, 0..1023
        int h = idx >> 2, part = idx & 3;       // vT tile decomposition
        const uint4* vg =
            (const uint4*)(vT + ((size_t)b * H_ + h) * S_ + j * 32 + part * 8);
#if HAVE_ASYNC_LDS
        cp16_async(krg + idx, kbl + idx);
        cp16_async(vg, vbl + idx);
#else
        kbl[idx] = krg[idx];
        vbl[idx] = *vg;
#endif
      }
#if HAVE_ASYNC_LDS
      wait_async0();
#endif
    }
    __syncthreads();  // staged tile visible to all 4 waves

    if (j <= jmax_self) {  // wave-uniform predicate: EXEC stays all-ones
      // scores: two 16x16 key sub-tiles, K/B fragments from LDS
      f32x8 sc[2];
#pragma unroll
      for (int t = 0; t < 2; ++t) {
        f32x8 s = {};
#pragma unroll
        for (int kk = 0; kk < 8; ++kk) {
          bf16x16 bf = load_frag(&kbuf[(t * 16 + lr) * 256 + kk * 32], c0);
          s = wmma_bf16(af[kk], bf, s);
        }
        sc[t] = s;
      }
      const float scale = 0.0625f;              // H^-0.5
#pragma unroll
      for (int t = 0; t < 2; ++t)
#pragma unroll
        for (int e = 0; e < 8; ++e) sc[t][e] *= scale;

      if (j * 32 + 31 > qt * 16) {              // diagonal region: causal mask
#pragma unroll
        for (int t = 0; t < 2; ++t) {
          int key = j * 32 + t * 16 + lr;       // D layout: col = lane&15
#pragma unroll
          for (int e = 0; e < 8; ++e) {
            int query = qt * 16 + e + 8 * half; // D layout: row = e + 8*half
            if (key > query) sc[t][e] = -1e30f;
          }
        }
      }

      // online softmax: row reductions across the 16 lanes of each half
      float alpha[8];
#pragma unroll
      for (int e = 0; e < 8; ++e) {
        float v = fmaxf(sc[0][e], sc[1][e]);
#pragma unroll
        for (int m = 1; m < 16; m <<= 1) v = fmaxf(v, __shfl_xor(v, m, 32));
        float mn = fmaxf(m_st[e], v);
        alpha[e] = __expf(m_st[e] - mn);
        m_st[e] = mn;
      }
#pragma unroll
      for (int e = 0; e < 8; ++e) {
        float p0 = __expf(sc[0][e] - m_st[e]);
        float p1 = __expf(sc[1][e] - m_st[e]);
        sc[0][e] = p0; sc[1][e] = p1;
        float rs = p0 + p1;
#pragma unroll
        for (int m = 1; m < 16; m <<= 1) rs += __shfl_xor(rs, m, 32);
        l_st[e] = l_st[e] * alpha[e] + rs;
      }
#pragma unroll
      for (int nt = 0; nt < 16; ++nt)
#pragma unroll
        for (int e = 0; e < 8; ++e) acc[nt][e] *= alpha[e];

      // relayout P (score-D layout) -> bf16 A-fragment via this wave's LDS slab.
      // Same-wave LDS ops are in-order; wave_barrier only blocks reordering.
#pragma unroll
      for (int t = 0; t < 2; ++t)
#pragma unroll
        for (int e = 0; e < 8; ++e)
          pbuf[wave][(e + 8 * half) * 32 + t * 16 + lr] = f2bf(sc[t][e]);
      __builtin_amdgcn_wave_barrier();
      bf16x16 pf = load_frag(&pbuf[wave][lr * 32], c0);

      // PV: B-fragment lanes = rows of vT tile (columns of V), contiguous keys
#pragma unroll
      for (int nt = 0; nt < 16; ++nt) {
        bf16x16 bf = load_frag(&vbuf[(nt * 16 + lr) * 32], c0);
        acc[nt] = wmma_bf16(pf, bf, acc[nt]);
      }
    }
  }

#pragma unroll
  for (int e = 0; e < 8; ++e) l_st[e] = 1.0f / l_st[e];
#pragma unroll
  for (int nt = 0; nt < 16; ++nt)
#pragma unroll
    for (int e = 0; e < 8; ++e) {
      int row = e + 8 * half;
      out[(size_t)(b * S_ + qt * 16 + row) * H_ + nt * 16 + lr] = acc[nt][e] * l_st[e];
    }
}

// ---------- launch ----------

extern "C" void kernel_launch(void* const* d_in, const int* in_sizes, int n_in,
                              void* d_out, int out_size, void* d_ws, size_t ws_size,
                              hipStream_t stream) {
  const float* x  = (const float*)d_in[0];
  const float* r  = (const float*)d_in[1];
  // d_in[2] = tril (causal mask implied analytically)
  const float* wq = (const float*)d_in[3];
  const float* wk = (const float*)d_in[4];
  const float* wv = (const float*)d_in[5];
  float* out = (float*)d_out;

  // workspace layout
  char* ws = (char*)d_ws;
  unsigned short* wbf = (unsigned short*)ws;                    // 3*256*256 bf16 = 384 KB
  float2* cs = (float2*)(ws + 3 * 65536 * 2);                   // 2048*128 float2 = 2 MB
  unsigned short* qr = (unsigned short*)(ws + 3 * 65536 * 2 + (size_t)S_ * 128 * 8);
  unsigned short* kr = qr + (size_t)NROW_ * H_;                 // each 8 MB bf16
  unsigned short* vT = kr + (size_t)NROW_ * H_;

  cvt_weights_k<<<768, 256, 0, stream>>>(wq, wk, wv, wbf);
  build_cs_k<<<S_, 128, 0, stream>>>(r, cs);
  proj_rope_k<<<NROW_ / 16, 128, 0, stream>>>(x, wbf, cs, qr, kr, vT);
  attn_k<<<B_ * (S_ / 16 / 4), 128, 0, stream>>>(qr, kr, vT, out);
}
